// MultiHeadAttention_42185168781349
// MI455X (gfx1250) — matrix operand order repaired and checked by reading, not compiled, not run on hardware
//
#include <hip/hip_runtime.h>

// ---------------------------------------------------------------------------
// MI455X / gfx1250: f16 WMMA (16x16x32, fp32 accum) fused MHA.
//   Sizes: B=4, S=1024, D_MODEL=64, NUM_HEADS=64 (head axis = inner index).
// ---------------------------------------------------------------------------

typedef _Float16 v16h __attribute__((ext_vector_type(16)));
typedef _Float16 h8   __attribute__((ext_vector_type(8)));
typedef float    v8f  __attribute__((ext_vector_type(8)));

#define SEQ   1024
#define DM    64      // d_model (GEMM K for projections, per-head feature dim)
#define NCH   4096    // d_model * num_heads

union V16U { v16h v; h8 h[2]; };

__device__ __forceinline__ v8f wmma16(v16h a, v16h b, v8f c) {
  // D = A(16x32 f16) * B(32x16 f16) + C(16x16 f32)
  return __builtin_amdgcn_wmma_f32_16x16x32_f16(false, a, false, b,
                                                (short)0, c, false, false);
}

// A-operand (16x32, row-major source, rows stride `stride` f16 elems).
// Lane layout (ISA 7.12.2): m = lane&15; lanes<16 hold K=k0+0..7 & k0+16..23,
// lanes>=16 hold K=k0+8..15 & k0+24..31 (two contiguous 16B chunks).
__device__ __forceinline__ v16h load_a(const _Float16* base, int stride,
                                       int k0, int lane) {
  int m = lane & 15, hf = lane >> 4;
  const _Float16* r = base + (size_t)m * stride + k0 + hf * 8;
  V16U u;
  u.h[0] = *(const h8*)(r);
  u.h[1] = *(const h8*)(r + 16);
  return u.v;
}

// B-operand (32x16): source is column-major, i.e. base[n*stride + k].
// Lane n=lane&15 holds column n; lanes<16 hold K=k0+0..7 & k0+16..23,
// lanes>=16 hold K=k0+8..15 & k0+24..31 (two contiguous 16B chunks).
__device__ __forceinline__ v16h load_b(const _Float16* base, int stride,
                                       int k0, int lane) {
  int n = lane & 15, hf = lane >> 4;
  const _Float16* r = base + (size_t)n * stride + k0 + hf * 8;
  V16U u;
  u.h[0] = *(const h8*)(r);
  u.h[1] = *(const h8*)(r + 16);
  return u.v;
}

// ---------------------------------------------------------------------------
__global__ void mha_cvt_f16(const float* __restrict__ in,
                            _Float16* __restrict__ out, int n) {
  int i = blockIdx.x * blockDim.x + threadIdx.x;
  if (i < n) out[i] = (_Float16)in[i];
}

// out[c*rows + r] = in[r*cols + c]  (fp32 -> f16 transpose)
__global__ void mha_transpose_f16(const float* __restrict__ in,
                                  _Float16* __restrict__ out,
                                  int rows, int cols) {
  int i = blockIdx.x * blockDim.x + threadIdx.x;
  if (i < rows * cols) {
    int r = i / cols, c = i % cols;
    out[(size_t)c * rows + r] = (_Float16)in[i];
  }
}

// ---------------------------------------------------------------------------
// One block per (b, head d). LDS: K_head[1024][64] + Vt_head[64][1024] f16
// (256 KB) + 8 per-wave staging tiles (16 KB) = 272 KB < 320 KB WGP LDS.
// 8 waves; each wave owns 8 s-tiles of 16 rows, flash-softmax over t.
// ---------------------------------------------------------------------------
__global__ __launch_bounds__(256) void mha_attn(
    const _Float16* __restrict__ xb,   // [B][S][64] f16
    const int* __restrict__ mask,      // [B][1][S][S]
    const _Float16* __restrict__ WqT,  // [4096][64] f16 (transposed)
    const _Float16* __restrict__ WkT,
    const _Float16* __restrict__ WvT,
    const float* __restrict__ bq, const float* __restrict__ bk,
    const float* __restrict__ bv,
    _Float16* __restrict__ O_all)      // [B*S][4096] f16, channel = f*64+d
{
  extern __shared__ _Float16 lds[];
  _Float16* K_lds  = lds;                    // [1024][64]
  _Float16* Vt_lds = lds + SEQ * DM;         // [64][1024]
  _Float16* stageA = lds + 2 * SEQ * DM;     // [8][16*64]

  const int head = blockIdx.x;               // 0..255
  const int b    = head >> 6;
  const int d    = head & 63;
  const int lane = threadIdx.x & 31;
  const int wave = threadIdx.x >> 5;
  const int ln   = lane & 15;
  const int hf   = lane >> 4;

  const _Float16* xrow = xb + (size_t)b * SEQ * DM;

  // ---- Stage K_head and Vt_head via WMMA: 2*64*4 = 512 D-tiles over 8 waves
  for (int idx = wave; idx < 512; idx += 8) {
    int kv = idx >> 8;               // 0 = K, 1 = V
    int tt = (idx >> 2) & 63;        // t-tile
    int hh = idx & 3;                // feature tile
    const _Float16* WT = kv ? WvT : WkT;
    const float*   bi = kv ? bv : bk;
    v16h a0 = load_a(xrow + (size_t)tt * 16 * DM, DM, 0, lane);
    v16h a1 = load_a(xrow + (size_t)tt * 16 * DM, DM, 32, lane);
    v16h b0 = load_b(WT + ((size_t)hh * 1024 + d) * 64, NCH, 0, lane);
    v16h b1 = load_b(WT + ((size_t)hh * 1024 + d) * 64, NCH, 32, lane);
    v8f c = {};
    c = wmma16(a0, b0, c);
    c = wmma16(a1, b1, c);
    float bias = bi[(hh * 16 + ln) * 64 + d];
    if (kv == 0) {
#pragma unroll
      for (int i = 0; i < 8; ++i)
        K_lds[(tt * 16 + i + 8 * hf) * DM + hh * 16 + ln] =
            (_Float16)(c[i] + bias);
    } else {
#pragma unroll
      for (int i = 0; i < 8; ++i)
        Vt_lds[(hh * 16 + ln) * SEQ + tt * 16 + i + 8 * hf] =
            (_Float16)(c[i] + bias);
    }
  }
  __syncthreads();

  _Float16* st = stageA + wave * (16 * 64);
  const int* mbase = mask + (size_t)b * SEQ * SEQ;

  for (int jj = 0; jj < 8; ++jj) {
    const int sbase = (wave * 8 + jj) * 16;

    // ---- Q tile (16x64) for this s-strip, staged so we can re-layout as A
    v16h qa0 = load_a(xrow + (size_t)sbase * DM, DM, 0, lane);
    v16h qa1 = load_a(xrow + (size_t)sbase * DM, DM, 32, lane);
#pragma unroll
    for (int hh = 0; hh < 4; ++hh) {
      v16h b0 = load_b(WqT + ((size_t)hh * 1024 + d) * 64, NCH, 0, lane);
      v16h b1 = load_b(WqT + ((size_t)hh * 1024 + d) * 64, NCH, 32, lane);
      v8f c = {};
      c = wmma16(qa0, b0, c);
      c = wmma16(qa1, b1, c);
      float bias = bq[(hh * 16 + ln) * 64 + d];
#pragma unroll
      for (int i = 0; i < 8; ++i)
        st[(i + 8 * hf) * 64 + hh * 16 + ln] = (_Float16)(c[i] + bias);
    }
    __syncthreads();
    v16h Aq0 = load_a(st, 64, 0, lane);
    v16h Aq1 = load_a(st, 64, 32, lane);
    __syncthreads();   // st is reused below for the P tile

    v8f O0 = {}, O1 = {}, O2 = {}, O3 = {};
    float m8[8], l8[8];
#pragma unroll
    for (int i = 0; i < 8; ++i) { m8[i] = -1e30f; l8[i] = 0.0f; }

    for (int t = 0; t < SEQ; t += 32) {
      // scores S[16, t..t+32) = Q(16x64) . K^T, fp32
      v8f S0 = {}, S1 = {};
      S0 = wmma16(Aq0, load_b(K_lds + (size_t)t * DM, DM, 0, lane), S0);
      S0 = wmma16(Aq1, load_b(K_lds + (size_t)t * DM, DM, 32, lane), S0);
      S1 = wmma16(Aq0, load_b(K_lds + (size_t)(t + 16) * DM, DM, 0, lane), S1);
      S1 = wmma16(Aq1, load_b(K_lds + (size_t)(t + 16) * DM, DM, 32, lane), S1);

#pragma unroll
      for (int i = 0; i < 8; ++i) {
        int srow = sbase + i + 8 * hf;
        float s0v = S0[i] * 0.125f;    // / sqrt(64)
        float s1v = S1[i] * 0.125f;
        if (mbase[(size_t)srow * SEQ + t + ln] == 0)      s0v = -1e9f;
        if (mbase[(size_t)srow * SEQ + t + 16 + ln] == 0) s1v = -1e9f;
        // row max across the 16 lanes of this half-wave
        float v = fmaxf(s0v, s1v);
        v = fmaxf(v, __shfl_xor(v, 1, 32));
        v = fmaxf(v, __shfl_xor(v, 2, 32));
        v = fmaxf(v, __shfl_xor(v, 4, 32));
        v = fmaxf(v, __shfl_xor(v, 8, 32));
        float nm = fmaxf(m8[i], v);
        float al = __expf(m8[i] - nm);
        float p0 = __expf(s0v - nm);
        float p1 = __expf(s1v - nm);
        float rs = p0 + p1;
        rs += __shfl_xor(rs, 1, 32);
        rs += __shfl_xor(rs, 2, 32);
        rs += __shfl_xor(rs, 4, 32);
        rs += __shfl_xor(rs, 8, 32);
        l8[i] = l8[i] * al + rs;
        m8[i] = nm;
        O0[i] *= al; O1[i] *= al; O2[i] *= al; O3[i] *= al;
        // stage P (16x32, row-major) for A-operand re-layout
        st[(i + 8 * hf) * 32 + ln]      = (_Float16)p0;
        st[(i + 8 * hf) * 32 + 16 + ln] = (_Float16)p1;
      }
      __syncthreads();
      v16h Ap = load_a(st, 32, 0, lane);
      O0 = wmma16(Ap, load_b(Vt_lds + t,              SEQ, 0, lane), O0);
      O1 = wmma16(Ap, load_b(Vt_lds + 16 * SEQ + t,   SEQ, 0, lane), O1);
      O2 = wmma16(Ap, load_b(Vt_lds + 32 * SEQ + t,   SEQ, 0, lane), O2);
      O3 = wmma16(Ap, load_b(Vt_lds + 48 * SEQ + t,   SEQ, 0, lane), O3);
      __syncthreads();
    }

    // ---- epilogue: O /= l, store to O_all in pre-concat order (f*64 + d)
#pragma unroll
    for (int i = 0; i < 8; ++i) {
      float inv = 1.0f / l8[i];
      size_t orow = ((size_t)b * SEQ + sbase + i + 8 * hf) * NCH;
      O_all[orow + (size_t)(ln)      * 64 + d] = (_Float16)(O0[i] * inv);
      O_all[orow + (size_t)(16 + ln) * 64 + d] = (_Float16)(O1[i] * inv);
      O_all[orow + (size_t)(32 + ln) * 64 + d] = (_Float16)(O2[i] * inv);
      O_all[orow + (size_t)(48 + ln) * 64 + d] = (_Float16)(O3[i] * inv);
    }
    __syncthreads();
  }
}

// ---------------------------------------------------------------------------
// Output projection: Y[4096, 64] = O_all[4096, 4096] @ Wd + bd (fp32 out).
// One wave per 16x16 tile: 256 M-tiles x 4 N-tiles = 1024 waves.
// ---------------------------------------------------------------------------
__global__ __launch_bounds__(256) void mha_out_gemm(
    const _Float16* __restrict__ O_all,   // [4096][4096]
    const _Float16* __restrict__ WdT,     // [64][4096]
    const float* __restrict__ bd,
    float* __restrict__ out)              // [4096][64]
{
  int lane = threadIdx.x & 31, ln = lane & 15, hf = lane >> 4;
  int wid  = blockIdx.x * 8 + (threadIdx.x >> 5);   // 0..1023
  int mt = wid >> 2, nt = wid & 3;
  const _Float16* Ab = O_all + (size_t)mt * 16 * NCH;
  const _Float16* Bb = WdT   + (size_t)nt * 16 * NCH;
  v8f acc = {};
  for (int k0 = 0; k0 < NCH; k0 += 32) {
    v16h a  = load_a(Ab, NCH, k0, lane);
    v16h bb = load_b(Bb, NCH, k0, lane);
    acc = wmma16(a, bb, acc);
  }
  float bias = bd[nt * 16 + ln];
#pragma unroll
  for (int i = 0; i < 8; ++i) {
    int row = mt * 16 + i + 8 * hf;
    out[(size_t)row * 64 + nt * 16 + ln] = acc[i] + bias;
  }
}

// ---------------------------------------------------------------------------
extern "C" void kernel_launch(void* const* d_in, const int* in_sizes, int n_in,
                              void* d_out, int out_size, void* d_ws,
                              size_t ws_size, hipStream_t stream) {
  const float* x    = (const float*)d_in[0];
  const int*   mask = (const int*)d_in[1];
  const float* Wq   = (const float*)d_in[2];
  const float* bq   = (const float*)d_in[3];
  const float* Wk   = (const float*)d_in[4];
  const float* bk   = (const float*)d_in[5];
  const float* Wv   = (const float*)d_in[6];
  const float* bv   = (const float*)d_in[7];
  const float* Wd   = (const float*)d_in[8];
  const float* bd   = (const float*)d_in[9];
  float* out = (float*)d_out;

  // ---- workspace carve (all chunks 512B-aligned by construction)
  char* w = (char*)d_ws;
  _Float16* xb   = (_Float16*)w; w += (size_t)4 * SEQ * DM * 2;     // 512 KB
  _Float16* WqT  = (_Float16*)w; w += (size_t)NCH * DM * 2;          // 512 KB
  _Float16* WkT  = (_Float16*)w; w += (size_t)NCH * DM * 2;
  _Float16* WvT  = (_Float16*)w; w += (size_t)NCH * DM * 2;
  _Float16* WdT  = (_Float16*)w; w += (size_t)NCH * DM * 2;
  _Float16* Oall = (_Float16*)w; w += (size_t)4 * SEQ * NCH * 2;     // 32 MB

  const int nx = 4 * SEQ * DM;        // 262144
  const int nw = DM * NCH;            // 262144
  mha_cvt_f16<<<(nx + 255) / 256, 256, 0, stream>>>(x, xb, nx);
  mha_transpose_f16<<<(nw + 255) / 256, 256, 0, stream>>>(Wq, WqT, DM, NCH);
  mha_transpose_f16<<<(nw + 255) / 256, 256, 0, stream>>>(Wk, WkT, DM, NCH);
  mha_transpose_f16<<<(nw + 255) / 256, 256, 0, stream>>>(Wv, WvT, DM, NCH);
  mha_transpose_f16<<<(nw + 255) / 256, 256, 0, stream>>>(Wd, WdT, NCH, DM);

  size_t shmem = (size_t)(2 * SEQ * DM + 8 * 16 * 64) * sizeof(_Float16); // 278528 B
  mha_attn<<<256, 256, shmem, stream>>>(xb, mask, WqT, WkT, WvT,
                                        bq, bk, bv, Oall);
  mha_out_gemm<<<128, 256, 0, stream>>>(Oall, WdT, bd, out);
}
